// Attention_34875134443712
// MI455X (gfx1250) — matrix operand order repaired and checked by reading, not compiled, not run on hardware
//
#include <hip/hip_runtime.h>

typedef __attribute__((ext_vector_type(16))) _Float16 v16h;
typedef __attribute__((ext_vector_type(8)))  _Float16 v8h;
typedef __attribute__((ext_vector_type(8)))  float    v8f;
typedef __attribute__((ext_vector_type(4)))  float    v4f;
typedef __attribute__((ext_vector_type(4)))  int      i32x4;
typedef __attribute__((ext_vector_type(8)))  int      i32x8;
typedef __attribute__((ext_vector_type(4)))  unsigned int u32x4;
typedef unsigned long long u64;

#define AS1 __attribute__((address_space(1)))
#define AS3 __attribute__((address_space(3)))

#if __has_builtin(__builtin_amdgcn_tensor_load_to_lds)
#define HAVE_TDM 1
#else
#define HAVE_TDM 0
#endif

__device__ __forceinline__ v8f wmma_f16(v16h a, v16h b, v8f c) {
  // 8 args: (neg_a, A, neg_b, B, c_mod, C, reuse_a, reuse_b)
  return __builtin_amdgcn_wmma_f32_16x16x32_f16(false, a, false, b, (short)0, c, false, false);
}

__device__ __forceinline__ void wait_async0() {
#if __has_builtin(__builtin_amdgcn_s_wait_asynccnt)
  __builtin_amdgcn_s_wait_asynccnt(0);
#else
  asm volatile("s_wait_asynccnt 0" ::: "memory");
#endif
}

__device__ __forceinline__ void wait_tensor0() {
#if __has_builtin(__builtin_amdgcn_s_wait_tensorcnt)
  __builtin_amdgcn_s_wait_tensorcnt(0);
#else
  asm volatile("s_wait_tensorcnt 0" ::: "memory");
#endif
}

__device__ __forceinline__ unsigned lds_off(const void* p) {
  return (unsigned)(u64)(AS3 const void*)p;     // LDS byte offset
}

__device__ __forceinline__ void async_b128(const void* g, void* l) {
#if __has_builtin(__builtin_amdgcn_global_load_async_to_lds_b128)
  __builtin_amdgcn_global_load_async_to_lds_b128((AS1 i32x4*)g, (AS3 i32x4*)l, 0, 0);
#else
  *(i32x4*)l = *(const i32x4*)(g);
#endif
}

// TDM 2-D tile load: tile_d0 x tile_d1 elements of size (1<<ds_code) bytes
// from a row-major tensor (row length tensor_d0, row stride stride0 elements)
// into LDS at lds_byte. D# layout per CDNA5 ISA 8.3/8.4.
__device__ __forceinline__ void tdm_load_2d(
    unsigned lds_byte, u64 gaddr, unsigned ds_code,
    unsigned tensor_d0, unsigned tensor_d1,
    unsigned tile_d0, unsigned tile_d1, u64 stride0)
{
  u32x4 g0;
  g0[0] = 1u;                                            // count=1
  g0[1] = lds_byte;                                      // lds_addr
  g0[2] = (unsigned)(gaddr & 0xffffffffu);               // global_addr[31:0]
  g0[3] = (unsigned)((gaddr >> 32) & 0x01ffffffu)        // global_addr[56:32]
        | (2u << 30);                                    // type=2 ("image")
  i32x8 g1;
  g1[0] = (int)(ds_code << 16);                          // data_size
  g1[1] = (int)((tensor_d0 & 0xffffu) << 16);            // tensor_dim0[15:0]
  g1[2] = (int)((tensor_d0 >> 16) | ((tensor_d1 & 0xffffu) << 16));
  g1[3] = (int)(((tensor_d1 >> 16) & 0xffffu) | (tile_d0 << 16));
  g1[4] = (int)(tile_d1 & 0xffffu);                      // tile_dim2 = 0
  g1[5] = (int)(stride0 & 0xffffffffu);                  // tensor_dim0_stride
  g1[6] = (int)((stride0 >> 32) & 0xffffu);              // dim1_stride = 0
  g1[7] = 0;
  i32x4 gz = {0, 0, 0, 0};
#if HAVE_TDM
#if __clang_major__ >= 23
  i32x8 gz8 = {0, 0, 0, 0, 0, 0, 0, 0};
  __builtin_amdgcn_tensor_load_to_lds(g0, g1, gz, gz, gz8, 0);
#else
  __builtin_amdgcn_tensor_load_to_lds(g0, g1, gz, gz, 0);
#endif
#endif
}

// ---------------------------------------------------------------------------
// Kernel 1: QKV = qkv_w (1536x512) @ x (512x1024) + bias, per batch.
// Block = 8 waves: 16(o) x 128(t) stripe. Per K-step the 32x128 f32 X chunk is
// staged into LDS with one TDM descriptor (async-LDS / direct copy fallback).
// Writes: q [b][h][t][e] f16 (pre-scaled by 1/8), k [b][h][t][e] f16,
//         vT [b][h][e][t] f16 (transposed for contiguous PV B-tiles).
// ---------------------------------------------------------------------------
__global__ __launch_bounds__(256) void qkv_kernel(
    const float* __restrict__ x, const float* __restrict__ qkv_w,
    const float* __restrict__ qkv_b,
    _Float16* __restrict__ q, _Float16* __restrict__ k,
    _Float16* __restrict__ vT)
{
  __shared__ float xs[32 * 128];               // 16 KB X chunk [c:32][t:128]
  const int tid  = threadIdx.x;
  const int lane = tid & 31;
  const int wid  = tid >> 5;
  // grid: 8 b * 96 mt * 8 ng = 6144 blocks
  const int ng = blockIdx.x & 7;
  const int mt = (blockIdx.x >> 3) % 96;
  const int b  = blockIdx.x / (8 * 96);
  const int o0  = mt * 16;
  const int tb0 = ng * 128;
  const int hi = lane >> 4, ln = lane & 15;
  const int tcol = wid * 16 + ln;              // this lane's t column (0..127)

  const float* wrow = qkv_w + (size_t)(o0 + ln) * 512;   // A row M = ln
  const float* xb   = x + (size_t)b * 512 * 1024;

  v8f acc = {};
  for (int k0 = 0; k0 < 512; k0 += 32) {
    __syncthreads();                           // WAR vs previous iteration
#if HAVE_TDM
    if (wid == 0) {
      tdm_load_2d(lds_off(xs), (u64)(const void*)(xb + (size_t)k0 * 1024 + tb0),
                  /*f32*/2, /*tensor_d0*/1024, /*tensor_d1*/512,
                  /*tile_d0*/128, /*tile_d1*/32, /*stride0*/1024);
      wait_tensor0();
    }
#else
    #pragma unroll
    for (int i = 0; i < 4; ++i) {
      const int fi = tid + i * 256;            // float4 index 0..1023
      const int row = fi >> 5;
      const int col = (fi & 31) * 4;
      async_b128(xb + (size_t)(k0 + row) * 1024 + tb0 + col, &xs[fi * 4]);
    }
    wait_async0();
#endif
    __syncthreads();

    // ---- A 16x32 tile from global (contiguous 8-float runs) ----
    v16h a;
    #pragma unroll
    for (int j = 0; j < 8; ++j) {
      a[j]     = (_Float16)wrow[k0 + hi * 16 + j];
      a[8 + j] = (_Float16)wrow[k0 + hi * 16 + 8 + j];
    }
    // ---- B 32x16 tile from LDS: col N = tcol, elem j -> K = hi*16+j ----
    v16h bm;
    #pragma unroll
    for (int j = 0; j < 16; ++j)
      bm[j] = (_Float16)xs[(hi * 16 + j) * 128 + tcol];
    acc = wmma_f16(a, bm, acc);
  }

  // C layout: elem v -> row M = v + hi*8, col N = tcol
  #pragma unroll
  for (int vv = 0; vv < 8; ++vv) {
    const int M = vv + hi * 8;
    const int o = o0 + M;                      // 0..1535
    const int t = tb0 + tcol;
    const float val = acc[vv] + qkv_b[o];
    const int sel = o >> 9;                    // 0=q 1=k 2=v
    const int cc = o & 511;
    const int head = cc >> 6, e = cc & 63;
    const int bh = b * 8 + head;
    if (sel == 0)
      q[((size_t)bh * 1024 + t) * 64 + e] = (_Float16)(val * 0.125f);
    else if (sel == 1)
      k[((size_t)bh * 1024 + t) * 64 + e] = (_Float16)val;
    else
      vT[((size_t)bh * 64 + e) * 1024 + t] = (_Float16)val;
  }
}

// ---------------------------------------------------------------------------
// Kernel 2: flash attention. 8 waves/block share one (b,head); each wave owns
// a 16-query tile. K/V chunks (32 keys) are TDM-staged into LDS once per block
// and consumed by all 8 waves; online softmax; O accum in 4x v8f (16x64).
// Writes ao as [b][t][c] f16 (c = head*64+e) for contiguous proj B-tiles.
// ---------------------------------------------------------------------------
__global__ __launch_bounds__(256) void attn_kernel(
    const _Float16* __restrict__ q, const _Float16* __restrict__ k,
    const _Float16* __restrict__ vT, _Float16* __restrict__ ao)
{
  __shared__ _Float16 ks [32 * 64];            // K chunk [key][emb]   (4 KB)
  __shared__ _Float16 vs_[64 * 32];            // V chunk [emb][key]   (4 KB)
  __shared__ _Float16 pbuf[8][16 * 32];        // per-wave P staging   (8 KB)
  const int lane = threadIdx.x & 31;
  const int wid  = threadIdx.x >> 5;
  const int tile = blockIdx.x * 8 + wid;       // 4096 tiles
  const int qt = tile & 63;
  const int bh = tile >> 6;                    // block-uniform: b*8 + head
  const int b = bh >> 3, head = bh & 7;
  const int t0 = qt * 16;
  const int hi = lane >> 4, ln = lane & 15;

  // Q A-tiles: emb 0..31 (qa0) and 32..63 (qa1); contiguous 8-half runs.
  v16h qa0, qa1;
  {
    const v8h* p = (const v8h*)(q + ((size_t)bh * 1024 + t0 + ln) * 64);
    v8h x0 = p[2 * hi], x1 = p[2 * hi + 1], x2 = p[4 + 2 * hi], x3 = p[5 + 2 * hi];
    #pragma unroll
    for (int j = 0; j < 8; ++j) {
      qa0[j] = x0[j]; qa0[8 + j] = x1[j];
      qa1[j] = x2[j]; qa1[8 + j] = x3[j];
    }
  }

  float mrow[8], lrow[8];
  #pragma unroll
  for (int i = 0; i < 8; ++i) { mrow[i] = -3.0e38f; lrow[i] = 0.f; }
  v8f oa0 = {}, oa1 = {}, oa2 = {}, oa3 = {};

  _Float16* myp = pbuf[wid];
  const _Float16* kbase = k + (size_t)bh * 1024 * 64;
  const _Float16* vbase = vT + (size_t)bh * 64 * 1024;

  for (int kb = 0; kb < 1024; kb += 32) {
    __syncthreads();                           // WAR vs previous chunk reads
#if HAVE_TDM
    if (wid == 0) {
      // K rows kb..kb+31 of [1024][64] -> ks[32][64]
      tdm_load_2d(lds_off(ks), (u64)(const void*)(kbase + (size_t)kb * 64),
                  /*f16*/1, 64, 1024, 64, 32, 64);
      // V cols kb..kb+31 of vT [64][1024] -> vs_[64][32]
      tdm_load_2d(lds_off(vs_), (u64)(const void*)(vbase + kb),
                  /*f16*/1, 1024, 64, 32, 64, 1024);
      wait_tensor0();
    }
#else
    {
      const int tid = threadIdx.x;
      const int r  = tid >> 3, c8 = (tid & 7) * 8;   // K: 32x64 halfs
      async_b128(kbase + (size_t)(kb + r) * 64 + c8, &ks[r * 64 + c8]);
      const int r2 = tid >> 2, c2 = (tid & 3) * 8;   // V: 64x32 halfs
      async_b128(vbase + (size_t)r2 * 1024 + kb + c2, &vs_[r2 * 32 + c2]);
      wait_async0();
    }
#endif
    __syncthreads();

    // ---- scores: two 16x16 tiles (keys local 0..15, 16..31), from LDS ----
    v16h k0a = *(const v16h*)(ks + (size_t)ln * 64 + hi * 16);        // emb 0..31
    v16h k0b = *(const v16h*)(ks + (size_t)ln * 64 + 32 + hi * 16);   // emb 32..63
    v16h k1a = *(const v16h*)(ks + (size_t)(16 + ln) * 64 + hi * 16);
    v16h k1b = *(const v16h*)(ks + (size_t)(16 + ln) * 64 + 32 + hi * 16);
    v8f s0 = {}, s1 = {};
    s0 = wmma_f16(qa0, k0a, s0); s0 = wmma_f16(qa1, k0b, s0);
    s1 = wmma_f16(qa0, k1a, s1); s1 = wmma_f16(qa1, k1b, s1);

    // ---- online softmax (row = vv + hi*8 lives in a 16-lane group) ----
    #pragma unroll
    for (int vv = 0; vv < 8; ++vv) {
      float cm = fmaxf(s0[vv], s1[vv]);
      cm = fmaxf(cm, __shfl_xor(cm, 1));
      cm = fmaxf(cm, __shfl_xor(cm, 2));
      cm = fmaxf(cm, __shfl_xor(cm, 4));
      cm = fmaxf(cm, __shfl_xor(cm, 8));
      const float mnew  = fmaxf(mrow[vv], cm);
      const float alpha = __expf(mrow[vv] - mnew);
      const float p0 = __expf(s0[vv] - mnew);
      const float p1 = __expf(s1[vv] - mnew);
      float rs = p0 + p1;
      rs += __shfl_xor(rs, 1); rs += __shfl_xor(rs, 2);
      rs += __shfl_xor(rs, 4); rs += __shfl_xor(rs, 8);
      lrow[vv] = lrow[vv] * alpha + rs;
      mrow[vv] = mnew;
      oa0[vv] *= alpha; oa1[vv] *= alpha; oa2[vv] *= alpha; oa3[vv] *= alpha;
      const int M = vv + hi * 8;
      myp[M * 32 + ln]      = (_Float16)p0;   // C-layout -> row-major LDS
      myp[M * 32 + 16 + ln] = (_Float16)p1;
    }
    asm volatile("s_wait_dscnt 0" ::: "memory");  // intra-wave LDS RAW

    // ---- reload P in A layout (16x32, K = keys of this chunk) ----
    v16h pa;
    {
      const v8h* pp = (const v8h*)(myp + ln * 32);
      v8h x0 = pp[2 * hi], x1 = pp[2 * hi + 1];
      #pragma unroll
      for (int j = 0; j < 8; ++j) { pa[j] = x0[j]; pa[8 + j] = x1[j]; }
    }

    // ---- O += P @ V : B[K=key][N=emb] from LDS vs_[emb][key] ----
    v16h vt;
    vt = *(const v16h*)(vs_ + (size_t)( 0 + ln) * 32 + hi * 16); oa0 = wmma_f16(pa, vt, oa0);
    vt = *(const v16h*)(vs_ + (size_t)(16 + ln) * 32 + hi * 16); oa1 = wmma_f16(pa, vt, oa1);
    vt = *(const v16h*)(vs_ + (size_t)(32 + ln) * 32 + hi * 16); oa2 = wmma_f16(pa, vt, oa2);
    vt = *(const v16h*)(vs_ + (size_t)(48 + ln) * 32 + hi * 16); oa3 = wmma_f16(pa, vt, oa3);
  }

  // ---- normalize and emit ao[b][t][c], c = head*64 + e ----
  const int c0 = head * 64;
  #pragma unroll
  for (int vv = 0; vv < 8; ++vv) {
    const int M = vv + hi * 8;
    const float inv = 1.0f / lrow[vv];
    _Float16* dst = ao + ((size_t)b * 1024 + t0 + M) * 512 + c0;
    dst[ 0 + ln] = (_Float16)(oa0[vv] * inv);
    dst[16 + ln] = (_Float16)(oa1[vv] * inv);
    dst[32 + ln] = (_Float16)(oa2[vv] * inv);
    dst[48 + ln] = (_Float16)(oa3[vv] * inv);
  }
}

// ---------------------------------------------------------------------------
// Kernel 3: out = proj_w (512x512) @ ao (512x1024) + bias, per batch (f32 out)
// ---------------------------------------------------------------------------
__global__ __launch_bounds__(256) void proj_kernel(
    const _Float16* __restrict__ ao, const float* __restrict__ proj_w,
    const float* __restrict__ proj_b, float* __restrict__ y)
{
  const int lane = threadIdx.x & 31;
  const int wid  = threadIdx.x >> 5;
  const int tile = blockIdx.x * 8 + wid;       // 8 * 32 * 64 = 16384 tiles
  const int nt = tile & 63;
  const int mt = (tile >> 6) & 31;
  const int b  = tile >> 11;
  const int o0 = mt * 16, t0 = nt * 16;
  const int hi = lane >> 4, ln = lane & 15;

  v8f acc = {};
  const float* wrow = proj_w + (size_t)(o0 + ln) * 512;
  const _Float16* brow = ao + ((size_t)b * 1024 + t0 + ln) * 512;

  for (int k0 = 0; k0 < 512; k0 += 32) {
    v16h a;
    #pragma unroll
    for (int j = 0; j < 8; ++j) {
      a[j]     = (_Float16)wrow[k0 + hi * 16 + j];
      a[8 + j] = (_Float16)wrow[k0 + hi * 16 + 8 + j];
    }
    // B[K=c][N=t]: col N = ln (t fixed), elem j -> c = k0+hi*16+j (contiguous)
    v16h bm = *(const v16h*)(brow + k0 + hi * 16);
    acc = wmma_f16(a, bm, acc);
  }

  #pragma unroll
  for (int vv = 0; vv < 8; ++vv) {
    const int o = o0 + vv + hi * 8;
    y[((size_t)b * 512 + o) * 1024 + t0 + ln] = acc[vv] + proj_b[o];
  }
}

// ---------------------------------------------------------------------------
extern "C" void kernel_launch(void* const* d_in, const int* in_sizes, int n_in,
                              void* d_out, int out_size, void* d_ws, size_t ws_size,
                              hipStream_t stream) {
  (void)in_sizes; (void)n_in; (void)out_size; (void)ws_size;
  const float* x      = (const float*)d_in[0];
  const float* qkv_w  = (const float*)d_in[1];
  const float* qkv_b  = (const float*)d_in[2];
  const float* proj_w = (const float*)d_in[3];
  const float* proj_b = (const float*)d_in[4];
  float* out = (float*)d_out;

  char* ws = (char*)d_ws;                       // 32 MB used, fully overwritten
  _Float16* q  = (_Float16*)(ws);                                  // 8 MB
  _Float16* k  = (_Float16*)(ws + (size_t)8  * 1024 * 1024);       // 8 MB
  _Float16* vT = (_Float16*)(ws + (size_t)16 * 1024 * 1024);       // 8 MB
  _Float16* ao = (_Float16*)(ws + (size_t)24 * 1024 * 1024);       // 8 MB

  qkv_kernel <<<6144, 256, 0, stream>>>(x, qkv_w, qkv_b, q, k, vT);
  attn_kernel<<< 512, 256, 0, stream>>>(q, k, vT, ao);
  proj_kernel<<<2048, 256, 0, stream>>>(ao, proj_w, proj_b, out);
}
